// DeepseekV4Indexer_89137751261354
// MI455X (gfx1250) — matrix operand order repaired and checked by reading, not compiled, not run on hardware
//
#include <hip/hip_runtime.h>
#include <hip/hip_bf16.h>

// ---------------------------------------------------------------------------
// Types for CDNA5 WMMA (wave32): v_wmma_f32_16x16x32_bf16
// ---------------------------------------------------------------------------
typedef __attribute__((ext_vector_type(16))) __bf16 bf16x16;
typedef __attribute__((ext_vector_type(8)))  float  f32x8;
typedef __attribute__((ext_vector_type(4))) unsigned int v4u32;
typedef __attribute__((ext_vector_type(8))) int v8i32;
typedef __attribute__((ext_vector_type(4))) int v4i32;

union FragBF { uint4 u[2]; bf16x16 v; };

static __device__ __forceinline__ unsigned pk2(float a, float b) {
  union { __bf16 h[2]; unsigned u; } x;
  x.h[0] = (__bf16)a; x.h[1] = (__bf16)b;
  return x.u;
}

// Problem constants
#define BB   2
#define SS   8192
#define HID  2048
#define NH   16
#define HD   128
#define RATE 32
#define NW   256          // S / RATE
#define QLR  1536
#define NPAD 768          // 528 weight rows padded to 3*256
#define MTOT (BB*SS)      // 16384

// ---------------------------------------------------------------------------
// CDNA5 async data movement feature detection
// ---------------------------------------------------------------------------
#if defined(__gfx1250__) && __has_builtin(__builtin_amdgcn_tensor_load_to_lds)
#define HAVE_TDM 1
#endif
#if defined(__gfx1250__) && __has_builtin(__builtin_amdgcn_global_load_async_to_lds_b128)
#define HAVE_ASYNC 1
#endif

typedef const __attribute__((address_space(1))) void gvoid_t;
typedef __attribute__((address_space(3))) void lvoid_t;

#if HAVE_TDM
// Issue one TDM 2D tile load: tile_d0 x tile_d1 elements (2 bytes each) from a
// row-major tensor with dim0 stride stride0 (elements) into packed LDS at lds_off.
static __device__ __forceinline__ void tdm_load_2d_to_lds(
    const void* gaddr, unsigned lds_off,
    unsigned tile_d0, unsigned tile_d1,
    unsigned td0, unsigned td1, unsigned long long stride0)
{
  const unsigned long long ga = (unsigned long long)(size_t)gaddr;
  v4u32 g0;
  g0[0] = 1u;                                               // count=1, user D#
  g0[1] = lds_off;                                          // lds_addr (bytes)
  g0[2] = (unsigned)(ga & 0xffffffffull);                   // global_addr[31:0]
  g0[3] = (unsigned)((ga >> 32) & 0x1ffffffull) | (2u << 30); // addr[56:32] | type=2
  v8i32 g1;
  g1[0] = 0x00010000;                                       // data_size=1 (2B/elem)
  g1[1] = (int)((td0 & 0xffffu) << 16);                     // tensor_dim0[15:0] @ [63:48]
  g1[2] = (int)(((td0 >> 16) & 0xffffu) | ((td1 & 0xffffu) << 16));
  g1[3] = (int)(((td1 >> 16) & 0xffffu) | ((tile_d0 & 0xffffu) << 16));
  g1[4] = (int)(tile_d1 & 0xffffu);                         // tile_dim1; tile_dim2=0
  g1[5] = (int)(unsigned)(stride0 & 0xffffffffull);         // dim0_stride lo32
  g1[6] = (int)(unsigned)((stride0 >> 32) & 0xffffull);     // dim0_stride hi16
  g1[7] = 0;
  v4i32 gz = {0, 0, 0, 0};
#if __has_include(<hip/amd_detail/amd_gfx1250_TDM.h>)
  v8i32 gz8 = {0, 0, 0, 0, 0, 0, 0, 0};
  __builtin_amdgcn_tensor_load_to_lds(g0, g1, gz, gz, gz8, 0);   // 6-arg toolchain
#else
  __builtin_amdgcn_tensor_load_to_lds(g0, g1, gz, gz, 0);        // 5-arg toolchain
#endif
}
#endif

// ---------------------------------------------------------------------------
// Kernel 0: pack weights to bf16.  Wc[768][2048] = [W_kv; W_gate; W_w; 0pad]
//           Wq[2048][1536] = bf16(W_qb)
// ---------------------------------------------------------------------------
__global__ void pack_weights(const float* __restrict__ Wkv, const float* __restrict__ Wg,
                             const float* __restrict__ Ww,  const float* __restrict__ Wqb,
                             __bf16* __restrict__ Wc, __bf16* __restrict__ Wq)
{
  const size_t stride = (size_t)gridDim.x * blockDim.x;
  const size_t i0 = (size_t)blockIdx.x * blockDim.x + threadIdx.x;
  const size_t n1 = (size_t)NPAD * HID;
  for (size_t j = i0; j < n1; j += stride) {
    int row = (int)(j >> 11), col = (int)(j & 2047);
    float v = 0.f;
    if (row < 256)      v = Wkv[(size_t)row * HID + col];
    else if (row < 512) v = Wg[(size_t)(row - 256) * HID + col];
    else if (row < 528) v = Ww[(size_t)(row - 512) * HID + col];
    Wc[j] = (__bf16)v;
  }
  const size_t n2 = (size_t)2048 * QLR;
  for (size_t j = i0; j < n2; j += stride) Wq[j] = (__bf16)Wqb[j];
}

// ---------------------------------------------------------------------------
// Kernel 1/2: bf16 WMMA GEMM.  C[M,N] = A[M,K](f32, converted inline) * B[N,K]^T
// Block tile 128x256, 8 waves of 64x64, KS=32, double-buffered LDS.
// B tile staged via TDM (tensor_load_to_lds) when available.
// ---------------------------------------------------------------------------
template <bool STORE_BF16>
__global__ __launch_bounds__(256, 1) void gemm_bf16_wmma(
    const float* __restrict__ A, const __bf16* __restrict__ Bw,
    void* __restrict__ Cout, int M, int N, int K)
{
  constexpr int BM = 128, BN = 256, KS = 32;
  __shared__ __bf16 sA[2][BM * KS];   // 16 KB
  __shared__ __bf16 sB[2][BN * KS];   // 32 KB
  const int tid  = threadIdx.x;
  const int wid  = tid >> 5, lane = tid & 31;
  const int m0   = blockIdx.y * BM;
  const int n0   = blockIdx.x * BN;
  const int waveM = (wid >> 2) * 64;
  const int waveN = (wid & 3) * 64;

  f32x8 acc[4][4] = {};

  const int arow_st = tid >> 1;           // 2 threads per A row
  const int aseg    = (tid & 1) * 16;     // 16 floats each
  const size_t aBase = (size_t)(m0 + arow_st) * K + aseg;
  const size_t bBase = (size_t)(n0 + tid) * K;  // 1 thread per B row (fallback path)
  (void)bBase;

  auto stageA = [&](int ks, int buf) {
    const float4* ap = (const float4*)(A + aBase + (size_t)ks * KS);
    float4 a0 = ap[0], a1 = ap[1], a2 = ap[2], a3 = ap[3];
    uint4 p0, p1;
    p0.x = pk2(a0.x, a0.y); p0.y = pk2(a0.z, a0.w);
    p0.z = pk2(a1.x, a1.y); p0.w = pk2(a1.z, a1.w);
    p1.x = pk2(a2.x, a2.y); p1.y = pk2(a2.z, a2.w);
    p1.z = pk2(a3.x, a3.y); p1.w = pk2(a3.z, a3.w);
    uint4* adst = (uint4*)&sA[buf][arow_st * KS + aseg];
    adst[0] = p0; adst[1] = p1;
  };

  auto loadB = [&](int ks, int buf) {
#if HAVE_TDM
    if (wid == 0) {
      const unsigned lds_off = (unsigned)(size_t)&sB[buf][0];  // flat LDS addr low bits = offset
      tdm_load_2d_to_lds(Bw + (size_t)n0 * K + (size_t)ks * KS, lds_off,
                         (unsigned)KS, (unsigned)BN,
                         (unsigned)K, (unsigned)N, (unsigned long long)K);
    }
#elif HAVE_ASYNC
    gvoid_t* gsrc = (gvoid_t*)(Bw + bBase + (size_t)ks * KS);
    lvoid_t* ldst = (lvoid_t*)&sB[buf][tid * KS];
    __builtin_amdgcn_global_load_async_to_lds_b128(gsrc, ldst, 0, 0);
    __builtin_amdgcn_global_load_async_to_lds_b128(gsrc, ldst, 16, 0);
    __builtin_amdgcn_global_load_async_to_lds_b128(gsrc, ldst, 32, 0);
    __builtin_amdgcn_global_load_async_to_lds_b128(gsrc, ldst, 48, 0);
#else
    const uint4* bp = (const uint4*)(Bw + bBase + (size_t)ks * KS);
    uint4 b0 = bp[0], b1 = bp[1], b2 = bp[2], b3 = bp[3];
    uint4* bdst = (uint4*)&sB[buf][tid * KS];
    bdst[0] = b0; bdst[1] = b1; bdst[2] = b2; bdst[3] = b3;
#endif
  };

  auto waitCopies = [&]() {
#if HAVE_TDM
    if (wid == 0) __builtin_amdgcn_s_wait_tensorcnt(0);
#elif HAVE_ASYNC
#if __has_builtin(__builtin_amdgcn_s_wait_asynccnt)
    __builtin_amdgcn_s_wait_asynccnt(0);
#else
    asm volatile("s_wait_asynccnt 0" ::: "memory");
#endif
#endif
  };

  auto mmaStep = [&](int buf) {
    FragBF af[4], bfrag[4];
    const int ar  = waveM + (lane & 15);
    const int klo = (lane >> 4) * 16;    // A: lanes>=16 hold K 16..31
    #pragma unroll
    for (int mt = 0; mt < 4; mt++) {
      const uint4* p = (const uint4*)&sA[buf][(ar + mt * 16) * KS + klo];
      af[mt].u[0] = p[0];     // K = klo .. klo+7
      af[mt].u[1] = p[1];     // K = klo+8 .. klo+15
    }
    const int bc = waveN + (lane & 15);
    const int kh = (lane >> 4) * 16;     // B: lanes>=16 hold K 16..31
    #pragma unroll
    for (int nt = 0; nt < 4; nt++) {
      const uint4* p = (const uint4*)&sB[buf][(bc + nt * 16) * KS + kh];
      bfrag[nt].u[0] = p[0]; bfrag[nt].u[1] = p[1];
    }
    #pragma unroll
    for (int mt = 0; mt < 4; mt++)
      #pragma unroll
      for (int nt = 0; nt < 4; nt++)
        acc[mt][nt] = __builtin_amdgcn_wmma_f32_16x16x32_bf16(
            false, af[mt].v, false, bfrag[nt].v, (short)0, acc[mt][nt], false, false);
  };

  const int nk = K / KS;
  stageA(0, 0);
  loadB(0, 0);
  waitCopies();
  __syncthreads();
  for (int ks = 0; ks < nk; ks++) {
    const int buf = ks & 1;
    if (ks + 1 < nk) { stageA(ks + 1, buf ^ 1); loadB(ks + 1, buf ^ 1); }
    mmaStep(buf);
    waitCopies();
    __syncthreads();
  }

  const int cr0 = m0 + waveM + (lane >> 4) * 8;
  const int cc0 = n0 + waveN + (lane & 15);
  #pragma unroll
  for (int mt = 0; mt < 4; mt++)
    #pragma unroll
    for (int nt = 0; nt < 4; nt++) {
      const int col = cc0 + nt * 16;
      const int row = cr0 + mt * 16;
      if constexpr (STORE_BF16) {
        __bf16* C = (__bf16*)Cout;
        #pragma unroll
        for (int v = 0; v < 8; v++) C[(size_t)(row + v) * N + col] = (__bf16)acc[mt][nt][v];
      } else {
        float* C = (float*)Cout;
        #pragma unroll
        for (int v = 0; v < 8; v++) C[(size_t)(row + v) * N + col] = acc[mt][nt][v];
      }
    }
}

// ---------------------------------------------------------------------------
// Kernel 3: windowed softmax pooling + RMSNorm + RoPE -> pooled bf16 [B][256][128]
// One block per (b, window); thread d = channel.
// kvg layout per token row (stride 768): [0:128)=K, [128:256)=V, [256:512)=gate, [512:528)=w
// ---------------------------------------------------------------------------
__global__ __launch_bounds__(128) void pool_kernel(
    const float* __restrict__ kvg, const float* __restrict__ pbias,
    const float* __restrict__ normw, __bf16* __restrict__ pooledB)
{
  const int b = blockIdx.x >> 8;
  const int n = blockIdx.x & 255;
  const int d = threadIdx.x;
  __shared__ float sG[64][128];
  __shared__ float sred[128];
  __shared__ float sp[128];

  for (int r = 0; r < 32; r++) {
    if (n > 0) {
      const size_t base = ((size_t)b * SS + (size_t)(n - 1) * RATE + r) * NPAD;
      sG[r][d] = kvg[base + 256 + d] + pbias[r * 256 + d];            // prev-window K gate
    } else {
      sG[r][d] = -__builtin_inff();
    }
    const size_t base2 = ((size_t)b * SS + (size_t)n * RATE + r) * NPAD;
    sG[32 + r][d] = kvg[base2 + 384 + d] + pbias[r * 256 + 128 + d];  // cur-window V gate
  }
  __syncthreads();

  float mx = -__builtin_inff();
  for (int r = 0; r < 64; r++) mx = fmaxf(mx, sG[r][d]);
  float se = 0.f, acc = 0.f;
  for (int r = 0; r < 32; r++) {
    const float e = __expf(sG[r][d] - mx);
    se += e;
    if (n > 0) {
      const size_t base = ((size_t)b * SS + (size_t)(n - 1) * RATE + r) * NPAD;
      acc += e * kvg[base + d];
    }
  }
  for (int r = 0; r < 32; r++) {
    const float e = __expf(sG[32 + r][d] - mx);
    se += e;
    const size_t base2 = ((size_t)b * SS + (size_t)n * RATE + r) * NPAD;
    acc += e * kvg[base2 + 128 + d];
  }
  float p = acc / se;

  sred[d] = p * p;
  __syncthreads();
  for (int off = 64; off > 0; off >>= 1) {
    if (d < off) sred[d] += sred[d + off];
    __syncthreads();
  }
  const float var = sred[0] * (1.f / 128.f);
  p = p * rsqrtf(var + 1e-6f) * normw[d];
  sp[d] = p;
  __syncthreads();

  float outv;
  if (d < 64) {
    outv = p;                          // nope section
  } else {
    const int j = d - 64, i = j >> 1;
    const float f = (float)(n * RATE) * powf(10000.f, -(float)i * (1.f / 32.f));
    const float rh = (j < 32) ? -sp[96 + j] : sp[32 + j];
    outv = p * cosf(f) + rh * sinf(f);
  }
  pooledB[((size_t)b * NW + n) * HD + d] = (__bf16)outv;
}

// ---------------------------------------------------------------------------
// Kernel 4: scores (WMMA) + weighted head reduction + exact stable top-64
// One block = 4 positions. Wave w: position w>>1, N-half (w&1)*128.
// ---------------------------------------------------------------------------
__global__ __launch_bounds__(256) void score_topk_kernel(
    const __bf16* __restrict__ qB, const __bf16* __restrict__ pooledB,
    const float* __restrict__ kvg, const int* __restrict__ posids,
    int* __restrict__ out)
{
  const int s0  = blockIdx.x * 4;          // flat position base (b*S+s)
  const int b   = s0 >> 13;
  const int tid = threadIdx.x;
  const int wid = tid >> 5, lane = tid & 31;
  __shared__ __bf16 sQ[64][128];           // 16 KB: 4 positions x 16 heads
  __shared__ float  sScore[4][256];
  __shared__ float  sW[4][16];

  // Stage Q with RoPE applied (bf16 in, f32 rope math, bf16 out)
  {
    const int r    = tid >> 2;             // 64 rows
    const int pidx = s0 + (r >> 4);
    const __bf16* qrow = qB + (size_t)pidx * 2048 + (size_t)(r & 15) * HD;
    const float pos = (float)posids[pidx];
    const int c0 = (tid & 3) * 32;
    for (int c = c0; c < c0 + 32; c++) {
      float v;
      if (c < 64) {
        v = (float)qrow[c];
      } else {
        const int j = c - 64, i = j >> 1;
        const float f = pos * powf(10000.f, -(float)i * (1.f / 32.f));
        const float x = (float)qrow[c];
        const float rh = (j < 32) ? -(float)qrow[96 + j] : (float)qrow[32 + j];
        v = x * cosf(f) + rh * sinf(f);
      }
      sQ[r][c] = (__bf16)v;
    }
  }
  if (tid < 64) {
    const int pp = tid >> 4, h = tid & 15;
    sW[pp][h] = kvg[(size_t)(s0 + pp) * NPAD + 512 + h] * 0.25f;   // * H^-0.5
  }
  __syncthreads();

  // scores = relu(Q x pooled^T) * HD^-0.5 ; pooled read straight from L2
  const int p     = wid >> 1;
  const int nhalf = (wid & 1) * 128;
  const __bf16* Pb = pooledB + (size_t)b * NW * HD;
  f32x8 acc[8] = {};
  #pragma unroll
  for (int kk = 0; kk < 4; kk++) {
    FragBF af;
    const int ar  = p * 16 + (lane & 15);
    const int klo = (lane >> 4) * 16;
    const uint4* pa = (const uint4*)&sQ[ar][kk * 32 + klo];
    af.u[0] = pa[0]; af.u[1] = pa[1];
    const int kh = (lane >> 4) * 16;
    #pragma unroll
    for (int nt = 0; nt < 8; nt++) {
      const int col = nhalf + nt * 16 + (lane & 15);
      const uint4* pb = (const uint4*)(Pb + (size_t)col * HD + kk * 32 + kh);
      FragBF bfrag; bfrag.u[0] = pb[0]; bfrag.u[1] = pb[1];
      acc[nt] = __builtin_amdgcn_wmma_f32_16x16x32_bf16(
          false, af.v, false, bfrag.v, (short)0, acc[nt], false, false);
    }
  }

  // index_scores[n] = sum_h relu(score)*HD^-0.5 * w[h]
  #pragma unroll
  for (int nt = 0; nt < 8; nt++) {
    float partial = 0.f;
    #pragma unroll
    for (int v = 0; v < 8; v++) {
      const float sc = fmaxf(acc[nt][v], 0.f) * 0.08838834764831843f;  // 128^-0.5
      const int h = (lane >> 4) * 8 + v;
      partial += sc * sW[p][h];
    }
    partial += __shfl_xor(partial, 16, 32);
    if (lane < 16) sScore[p][nhalf + nt * 16 + lane] = partial;
  }
  __syncthreads();

  // Top-64 of 256: iterative wave argmax, ties -> lower index (matches lax.top_k)
  if (wid < 4) {
    float v[8];
    const int pw = wid;
    #pragma unroll
    for (int k = 0; k < 8; k++) v[k] = sScore[pw][lane * 8 + k];
    int* o = out + (size_t)(s0 + pw) * 64;
    for (int it = 0; it < 64; it++) {
      float bv = -__builtin_inff(); int bi = 0x7fffffff;
      #pragma unroll
      for (int k = 0; k < 8; k++) {
        const int ix = lane * 8 + k;
        if (v[k] > bv || (v[k] == bv && ix < bi)) { bv = v[k]; bi = ix; }
      }
      #pragma unroll
      for (int off = 16; off > 0; off >>= 1) {
        const float ov = __shfl_xor(bv, off, 32);
        const int   oi = __shfl_xor(bi, off, 32);
        if (ov > bv || (ov == bv && oi < bi)) { bv = ov; bi = oi; }
      }
      if ((bi >> 3) == lane) v[bi & 7] = -__builtin_inff();
      if (lane == 0) o[it] = bi;
    }
  }
}

// ---------------------------------------------------------------------------
extern "C" void kernel_launch(void* const* d_in, const int* in_sizes, int n_in,
                              void* d_out, int out_size, void* d_ws, size_t ws_size,
                              hipStream_t stream)
{
  const float* hidden = (const float*)d_in[0];
  const float* qres   = (const float*)d_in[1];
  const float* Wkv    = (const float*)d_in[2];
  const float* Wgate  = (const float*)d_in[3];
  const float* pbias  = (const float*)d_in[4];
  const float* normw  = (const float*)d_in[5];
  const float* Wqb    = (const float*)d_in[6];
  const float* Ww     = (const float*)d_in[7];
  const int*   posids = (const int*)d_in[8];
  int* out = (int*)d_out;

  char* ws = (char*)d_ws;
  // workspace layout (bytes)
  __bf16* WcB     = (__bf16*)(ws);                          //  768*2048*2 = 3,145,728
  __bf16* WqbB    = (__bf16*)(ws + 3145728);                // 2048*1536*2 = 6,291,456
  float*  kvg     = (float*)(ws + 9437184);                 // 16384*768*4 = 50,331,648
  __bf16* qBf     = (__bf16*)(ws + 59768832);               // 16384*2048*2 = 67,108,864
  __bf16* pooledB = (__bf16*)(ws + 126877696);              // 2*256*128*2 = 131,072

  pack_weights<<<dim3(4096), dim3(256), 0, stream>>>(Wkv, Wgate, Ww, Wqb, WcB, WqbB);

  // kv|gate|w projection: C f32 [16384, 768]
  gemm_bf16_wmma<false><<<dim3(NPAD / 256, MTOT / 128), dim3(256), 0, stream>>>(
      hidden, WcB, (void*)kvg, MTOT, NPAD, HID);

  // q projection: C bf16 [16384, 2048]
  gemm_bf16_wmma<true><<<dim3(2048 / 256, MTOT / 128), dim3(256), 0, stream>>>(
      qres, WqbB, (void*)qBf, MTOT, 2048, QLR);

  pool_kernel<<<dim3(BB * NW), dim3(128), 0, stream>>>(kvg, pbias, normw, pooledB);

  score_topk_kernel<<<dim3(MTOT / 4), dim3(256), 0, stream>>>(qBf, pooledB, kvg, posids, out);
}
